// ModeT_67714454389079
// MI455X (gfx1250) — matrix operand order repaired and checked by reading, not compiled, not run on hardware
//
#include <hip/hip_runtime.h>

typedef __attribute__((ext_vector_type(16))) _Float16 v16h;
typedef __attribute__((ext_vector_type(8)))  float    v8f;

namespace {
constexpr int Bc = 2, Hc = 48, Wc = 48, Tc = 48;
constexpr int DIMc = 384, NHc = 6, HDc = 64;
constexpr int TILE_Z = 14;   // valid output rows per 16-row WMMA tile (band p+dl <= 15)
constexpr int NZT    = 4;    // ceil(48/14)
constexpr int WAVES  = 4;    // waves per block
}

__global__ __launch_bounds__(WAVES * 32)
void natten3d_wmma_kernel(const float* __restrict__ q,
                          const float* __restrict__ k,
                          const float* __restrict__ rpb,
                          float* __restrict__ out)
{
    __shared__ float lds[WAVES][9 * 16 * 16];   // 9 D-tiles per wave

    const int lane = threadIdx.x & 31;
    const int wid  = threadIdx.x >> 5;

    // ---- decompose flattened tile index: (b, x, y, head, ztile) ----
    int t = blockIdx.x * WAVES + wid;
    const int tz = t & 3;        t >>= 2;
    const int n  = t % NHc;      t /= NHc;
    const int y  = t % Wc;       t /= Wc;
    const int x  = t % Hc;       t /= Hc;
    const int b  = t;
    const int z0 = tz * TILE_Z;

    const int col = lane & 15;   // A: row M ; B: column N ; D: column N
    const int hi  = lane >> 4;   // selects K sub-range per ISA layout

    // Uniform (per-wave) row base in floats for the center voxel's head slice.
    const size_t ubase = (((size_t)(b * Hc + x) * Wc + y) * Tc) * DIMc + n * HDc;
    const float* qu = q + ubase;
    const float* ku = k + ubase;

    // ------------------------------------------------------------------
    // Load A = Q tile (16 positions x 64 dims), f32 -> f16, pre-scaled.
    // Unconditional loads from a z-CLAMPED address; rows past the end are
    // garbage but never read back (softmax only covers valid positions).
    // A 16x32 layout: lane<16: halves 0..7 -> K = hi*8+0..7,
    //                           halves 8..15 -> K = 16+hi*8+0..7
    // ------------------------------------------------------------------
    v16h a[2];
    {
        const int zq    = (z0 + col < Tc) ? (z0 + col) : (Tc - 1);
        const int qoffs = zq * DIMc;               // per-lane 32-bit offset
        const float s = 0.125f;                    // HD^-0.5
        #pragma unroll
        for (int c = 0; c < 2; ++c) {              // K-chunks of 32
            #pragma unroll
            for (int g = 0; g < 2; ++g) {          // halves g*8 .. g*8+7
                const int kb = c * 32 + g * 16 + hi * 8;
                float4 f0 = *(const float4*)(qu + qoffs + kb);
                float4 f1 = *(const float4*)(qu + qoffs + kb + 4);
                a[c][g * 8 + 0] = (_Float16)(f0.x * s);
                a[c][g * 8 + 1] = (_Float16)(f0.y * s);
                a[c][g * 8 + 2] = (_Float16)(f0.z * s);
                a[c][g * 8 + 3] = (_Float16)(f0.w * s);
                a[c][g * 8 + 4] = (_Float16)(f1.x * s);
                a[c][g * 8 + 5] = (_Float16)(f1.y * s);
                a[c][g * 8 + 6] = (_Float16)(f1.z * s);
                a[c][g * 8 + 7] = (_Float16)(f1.w * s);
            }
        }
    }

    // ------------------------------------------------------------------
    // Banded GEMM per neighbor offset (di,dj):  D = Q(16x64) x K(64x16).
    // K columns are z' = z0-1 .. z0+14. Out-of-bounds rows/columns are NOT
    // zeroed here: addresses are clamped in-bounds and the boundary mask is
    // applied once at softmax-read time (OOB logit == 0 in the reference).
    // ------------------------------------------------------------------
    const int zc  = z0 - 1 + col;                  // this lane's k z-index
    const int zcl = (zc < 0) ? 0 : ((zc >= Tc) ? (Tc - 1) : zc);
    const int koffs = zcl * DIMc;                  // per-lane 32-bit offset

    const bool xok0 = (x >= 1),      yok0 = (y >= 1);
    const bool xok2 = (x + 1 < Hc),  yok2 = (y + 1 < Wc);
    // Uniform clamp offsets (floats) for the 3 neighbor rows per axis.
    const int xs = Wc * Tc * DIMc, ys = Tc * DIMc;
    const int xoff[3] = { xok0 ? -xs : 0, 0, xok2 ? xs : 0 };
    const int yoff[3] = { yok0 ? -ys : 0, 0, yok2 ? ys : 0 };

    float* lwave = lds[wid];
    const int dstoff = (hi * 8) * 16 + col;        // D: M = r + 8*hi, N = col

    #pragma unroll
    for (int t9 = 0; t9 < 9; ++t9) {
        const int di = t9 / 3, dj = t9 % 3;
        const float* kt = ku + xoff[di] + yoff[dj];   // uniform per tile

        v8f acc = (v8f){0.f, 0.f, 0.f, 0.f, 0.f, 0.f, 0.f, 0.f};
        #pragma unroll
        for (int c = 0; c < 2; ++c) {
            // B 32x16 layout: col=lane%16, halves 0..7 -> K = hi*8+0..7,
            //                 halves 8..15 -> K = 16+hi*8+0..7
            const int kb = c * 32 + hi * 8;
            float4 g0 = *(const float4*)(kt + koffs + kb);
            float4 g1 = *(const float4*)(kt + koffs + kb + 4);
            float4 g2 = *(const float4*)(kt + koffs + kb + 16);
            float4 g3 = *(const float4*)(kt + koffs + kb + 20);
            v16h bm;
            bm[0]  = (_Float16)g0.x; bm[1]  = (_Float16)g0.y;
            bm[2]  = (_Float16)g0.z; bm[3]  = (_Float16)g0.w;
            bm[4]  = (_Float16)g1.x; bm[5]  = (_Float16)g1.y;
            bm[6]  = (_Float16)g1.z; bm[7]  = (_Float16)g1.w;
            bm[8]  = (_Float16)g2.x; bm[9]  = (_Float16)g2.y;
            bm[10] = (_Float16)g2.z; bm[11] = (_Float16)g2.w;
            bm[12] = (_Float16)g3.x; bm[13] = (_Float16)g3.y;
            bm[14] = (_Float16)g3.z; bm[15] = (_Float16)g3.w;

            acc = __builtin_amdgcn_wmma_f32_16x16x32_f16(
                /*neg_a=*/false, a[c], /*neg_b=*/false, bm,
                /*c_mod=*/(short)0, acc, /*reuse_a=*/false, /*reuse_b=*/false);
        }

        // Spill D tile to this wave's LDS region.
        float* lt = lwave + t9 * 256 + dstoff;
        #pragma unroll
        for (int r = 0; r < 8; ++r)
            lt[r * 16] = acc[r];
    }

    __syncthreads();   // orders the per-wave LDS writes before cross-lane reads

    // ------------------------------------------------------------------
    // Softmax + projection onto the fixed offset grid v[o] = (di-1,dj-1,dl-1).
    // Lane p handles position z0+p. Boundary masking happens HERE:
    // OOB neighbors get logit 0 (dot product with zero padding) + rpb.
    // ------------------------------------------------------------------
    if (lane < TILE_Z && (z0 + lane) < Tc) {
        const int p = lane;
        const float* lw = lds[wid];
        const float* rp = rpb + n * 27;

        const bool xv[3] = { xok0, true, xok2 };
        const bool yv[3] = { yok0, true, yok2 };

        float lg[27];
        float mx = -3.4e38f;
        #pragma unroll
        for (int o = 0; o < 27; ++o) {
            const int di = o / 9, dj = (o % 9) / 3, dl = o % 3;
            const bool zv = (unsigned)(z0 + p + dl - 1) < (unsigned)Tc;
            const float raw = (xv[di] && yv[dj] && zv)
                                ? lw[(di * 3 + dj) * 256 + p * 16 + (p + dl)]
                                : 0.f;
            const float v_ = raw + rp[o];
            lg[o] = v_;
            mx = fmaxf(mx, v_);
        }
        float den = 0.f, n0 = 0.f, n1 = 0.f, n2 = 0.f;
        #pragma unroll
        for (int o = 0; o < 27; ++o) {
            const int di = o / 9, dj = (o % 9) / 3, dl = o % 3;
            const float e = __expf(lg[o] - mx);
            den += e;
            n0 += e * (float)(di - 1);
            n1 += e * (float)(dj - 1);
            n2 += e * (float)(dl - 1);
        }
        const float inv = 1.0f / den;

        const size_t per = (size_t)Hc * Wc * Tc;   // 110592
        float* ob = out + ((size_t)b * NHc * 3 + n * 3) * per
                        + ((size_t)x * Wc + y) * Tc + (z0 + p);
        ob[0]       = n0 * inv;
        ob[per]     = n1 * inv;
        ob[2 * per] = n2 * inv;
    }
}

extern "C" void kernel_launch(void* const* d_in, const int* in_sizes, int n_in,
                              void* d_out, int out_size, void* d_ws, size_t ws_size,
                              hipStream_t stream) {
    const float* q   = (const float*)d_in[0];
    const float* k   = (const float*)d_in[1];
    const float* rpb = (const float*)d_in[2];
    // d_in[3] (offset grid v) is reconstructed analytically in-kernel.
    float* out = (float*)d_out;

    const int totalTiles = Bc * Hc * Wc * NHc * NZT;   // 110592, divisible by WAVES
    dim3 grid(totalTiles / WAVES);
    dim3 block(WAVES * 32);
    hipLaunchKernelGGL(natten3d_wmma_kernel, grid, block, 0, stream, q, k, rpb, out);
}
